// PointerNetwork_29171417874809
// MI455X (gfx1250) — matrix operand order repaired and checked by reading, not compiled, not run on hardware
//
#include <hip/hip_runtime.h>
#include <hip/hip_bf16.h>
#include <math.h>

// Problem dims (fixed by reference)
#define BATCH 512
#define SEQ   128
#define EMB   128
#define HID   256
#define KLSTM (EMB + HID)   // 384
#define G4    (4 * HID)     // 1024
#define NEGV  -100000.0f

typedef __bf16 bf16_t;
typedef __attribute__((ext_vector_type(16))) __bf16 v16bf;
typedef __attribute__((ext_vector_type(8)))  float  v8f;
typedef __attribute__((ext_vector_type(4)))  unsigned int v4u;
typedef __attribute__((ext_vector_type(8)))  int v8i;
typedef __attribute__((ext_vector_type(4)))  int v4i;

__device__ __forceinline__ float sigf(float x) { return 1.0f / (1.0f + __expf(-x)); }

// ---------------------------------------------------------------------------
// Generic bf16 WMMA GEMM:  out[M,N] = A[M,K] * W[N,K]^T + bias[N]
// Block = 256 threads (8 waves). Block tile = 64(M) x 128(N).
// The Tensor Data Mover DMAs the 64 x K A-slab into LDS (wave 0 issues the
// D#, waits on TENSORcnt, workgroup barrier releases all waves). Each wave
// owns one 16-wide N column and 4 M-subtiles: per K-step it loads one B
// fragment from global and reuses it for 4 WMMAs against 4 A fragments,
// accumulating into 4 independent v8f accumulators.
// ---------------------------------------------------------------------------
__global__ __launch_bounds__(256) void wmma_gemm_bf16(
    const bf16_t* __restrict__ A, const bf16_t* __restrict__ W,
    const float* __restrict__ bias, float* __restrict__ outF,
    bf16_t* __restrict__ outB, int M, int N, int K)
{
  __shared__ __attribute__((aligned(16))) bf16_t sA[64 * KLSTM];
  const int tid = threadIdx.x;
  const int m0  = blockIdx.y * 64;

#if __has_builtin(__builtin_amdgcn_tensor_load_to_lds)
  // --- TDM staging: one 2D tile descriptor, 64 rows x K cols of bf16 ---
  if (tid < 32) {
    const unsigned lds_base = (unsigned)(size_t)(void*)sA;   // LDS addr = low 32b
    const unsigned long long ga =
        (unsigned long long)(const void*)(A + (size_t)m0 * K);
    v4u g0;
    g0.x = 1u;                                      // count=1 (valid descriptor)
    g0.y = lds_base;                                // lds_addr [63:32]
    g0.z = (unsigned)ga;                            // global_addr [95:64]
    g0.w = (unsigned)((ga >> 32) & 0x01FFFFFFu)     // global_addr [120:96]
         | 0x80000000u;                             // type=2 ("image") [127:126]
    v8i g1;
    g1[0] = (int)(1u << 16);                        // data_size=1 -> 2 bytes
    g1[1] = (int)(((unsigned)K & 0xFFFFu) << 16);   // tensor_dim0[15:0] @ bits63:48
    g1[2] = (int)((((unsigned)K >> 16) & 0xFFFFu)   // tensor_dim0[31:16]
         | (64u << 16));                            // tensor_dim1[15:0] = 64 rows
    g1[3] = (int)(((unsigned)K & 0xFFFFu) << 16);   // tile_dim0 = K @ bits127:112
    g1[4] = 64;                                     // tile_dim1=64, tile_dim2=0
    g1[5] = K;                                      // tensor_dim0_stride[31:0]
    g1[6] = 0;                                      // stride hi, dim1_stride lo
    g1[7] = 0;
    const v4i gz4 = {0, 0, 0, 0};                   // 2D tensor: groups 2/3 unused
    const v8i gz8 = {0, 0, 0, 0, 0, 0, 0, 0};       // extra group (clang-23 form)
    __builtin_amdgcn_tensor_load_to_lds(g0, g1, gz4, gz4, gz8, 0);
    __builtin_amdgcn_s_wait_tensorcnt(0);
  }
  __syncthreads();
#else
  // Fallback: cooperative 128-bit staging of the A slab [64 x K] into LDS.
  const int elems = 64 * K;                         // K is a multiple of 8
  for (int idx = tid * 8; idx < elems; idx += 256 * 8) {
    int row = idx / K, col = idx % K;
    *(uint4*)&sA[idx] = *(const uint4*)&A[(size_t)(m0 + row) * K + col];
  }
  __syncthreads();
#endif

  const int wave = tid >> 5;
  const int lane = tid & 31;
  const int n0   = blockIdx.x * 128 + wave * 16;
  const int half = lane >> 4;        // 0: lanes 0-15, 1: lanes 16-31
  const int nloc = lane & 15;
  const int row  = nloc;             // A-fragment row within a 16-row subtile
  const int kh   = half * 8;         // A K-phase per ISA 16-bit A layout
  const int n    = n0 + nloc;

  v8f acc[4];
  #pragma unroll
  for (int mt = 0; mt < 4; ++mt) acc[mt] = (v8f){0.f,0.f,0.f,0.f,0.f,0.f,0.f,0.f};

  for (int k0 = 0; k0 < K; k0 += 32) {
    // B fragment (shared by 4 M-subtiles): K = 8*half+0..7 and 16+8*half+0..7 for column n
    union { uint4 q[2]; v16bf v; } ub;
    const bf16_t* wp = &W[(size_t)n * K + k0 + half * 8];
    __builtin_prefetch(wp + 32, 0, 0);
    ub.q[0] = *(const uint4*)wp;
    ub.q[1] = *(const uint4*)(wp + 16);
    #pragma unroll
    for (int mt = 0; mt < 4; ++mt) {
      union { uint4 q[2]; v16bf v; } ua;
      const int arow = mt * 16 + row;
      ua.q[0] = *(const uint4*)&sA[arow * K + k0 + kh];
      ua.q[1] = *(const uint4*)&sA[arow * K + k0 + kh + 16];
      acc[mt] = __builtin_amdgcn_wmma_f32_16x16x32_bf16(
          /*neg_a=*/false, ua.v, /*neg_b=*/false, ub.v,
          /*c_mod=*/(short)0, acc[mt], /*reuse_a=*/false, /*reuse_b=*/false);
    }
  }

  const float bv = bias ? bias[n] : 0.0f;
  #pragma unroll
  for (int mt = 0; mt < 4; ++mt) {
    #pragma unroll
    for (int r = 0; r < 8; ++r) {
      // C/D layout: VGPR r, lanes 0-15 => M=r ; lanes 16-31 => M=r+8
      const int m = m0 + mt * 16 + r + half * 8;
      const float v = acc[mt][r] + bv;
      const size_t off = (size_t)m * N + n;
      if (outF) outF[off] = v;
      if (outB) outB[off] = (bf16_t)v;
    }
  }
}

// ---------------------------------------------------------------------------
// Setup / conversion kernels
// ---------------------------------------------------------------------------
__global__ void kcvt(const float* __restrict__ s, bf16_t* __restrict__ d, int n) {
  int i = blockIdx.x * 256 + threadIdx.x;
  if (i < n) d[i] = (bf16_t)s[i];
}

// Build combined LSTM weight [4H x (E+H)] bf16 = [Wih | Whh]
__global__ void kcombine(const float* __restrict__ Wih, const float* __restrict__ Whh,
                         bf16_t* __restrict__ d) {
  int i = blockIdx.x * 256 + threadIdx.x;
  if (i >= G4 * KLSTM) return;
  int r = i / KLSTM, col = i % KLSTM;
  float v = (col < EMB) ? Wih[r * EMB + col] : Whh[r * HID + (col - EMB)];
  d[i] = (bf16_t)v;
}

__global__ void kbias2(const float* __restrict__ a, const float* __restrict__ b,
                       float* __restrict__ d, int n) {
  int i = blockIdx.x * 256 + threadIdx.x;
  if (i < n) d[i] = a[i] + b[i];
}

// emb[b,s,e] = x[b,s,0]*W[0,e] + x[b,s,1]*W[1,e]  (stored bf16)
__global__ void kembed(const float* __restrict__ x, const float* __restrict__ ew,
                       bf16_t* __restrict__ emb) {
  int i = blockIdx.x * 256 + threadIdx.x;
  if (i >= BATCH * SEQ * EMB) return;
  int e = i % EMB, bs = i / EMB;
  float v = x[bs * 2 + 0] * ew[e] + x[bs * 2 + 1] * ew[EMB + e];
  emb[i] = (bf16_t)v;
}

__global__ void kinit(float* __restrict__ h, float* __restrict__ c,
                      bf16_t* __restrict__ din, int* __restrict__ mask,
                      const float* __restrict__ dec_start) {
  int i = blockIdx.x * 256 + threadIdx.x;
  if (i < BATCH * HID) { h[i] = 0.0f; c[i] = 0.0f; }
  if (i < BATCH * EMB) din[i] = (bf16_t)dec_start[i % EMB];
  if (i < BATCH * SEQ) mask[i] = 0;
}

// Pack A = [x_t | h] as bf16 [B, 384]
__global__ void kpack(const bf16_t* __restrict__ xsrc, int t, int xIsEmb,
                      const float* __restrict__ h, bf16_t* __restrict__ A) {
  int i = blockIdx.x * 256 + threadIdx.x;
  if (i >= BATCH * KLSTM) return;
  int b = i / KLSTM, col = i % KLSTM;
  bf16_t v;
  if (col < EMB)
    v = xIsEmb ? xsrc[((size_t)b * SEQ + t) * EMB + col] : xsrc[b * EMB + col];
  else
    v = (bf16_t)h[b * HID + (col - EMB)];
  A[i] = v;
}

// LSTM pointwise: gate order i,f,g,o
__global__ void klstm(const float* __restrict__ gates, float* __restrict__ h,
                      float* __restrict__ c, bf16_t* __restrict__ hbf,
                      bf16_t* __restrict__ encout, int t) {
  int i = blockIdx.x * 256 + threadIdx.x;
  if (i >= BATCH * HID) return;
  int b = i / HID, hh = i % HID;
  const float* g = gates + (size_t)b * G4;
  float ig = sigf(g[hh]);
  float fg = sigf(g[HID + hh]);
  float gg = tanhf(g[2 * HID + hh]);
  float og = sigf(g[3 * HID + hh]);
  float cn = fg * c[i] + ig * gg;
  float hn = og * tanhf(cn);
  c[i] = cn; h[i] = hn; hbf[i] = (bf16_t)hn;
  if (encout) encout[((size_t)b * SEQ + t) * HID + hh] = (bf16_t)hn;
}

// logits[b,s] = V . tanh(ql[b,:] + ref[b,s,:]) ; pointer mode: 10*tanh(.)
__global__ __launch_bounds__(256) void kattlog(
    const float* __restrict__ ql, const bf16_t* __restrict__ ref,
    const float* __restrict__ V, const int* __restrict__ mask,
    float* __restrict__ logits, int pointer_mode) {
  int b = blockIdx.x, tid = threadIdx.x, wave = tid >> 5, lane = tid & 31;
  for (int s = wave; s < SEQ; s += 8) {
    const bf16_t* rp = ref + ((size_t)b * SEQ + s) * HID;
    float part = 0.0f;
    for (int hh = lane; hh < HID; hh += 32)
      part += V[hh] * tanhf(ql[b * HID + hh] + (float)rp[hh]);
    for (int off = 16; off; off >>= 1) part += __shfl_xor(part, off);
    if (lane == 0) {
      float lg = pointer_mode ? 10.0f * tanhf(part) : part;
      if (mask[b * SEQ + s]) lg = NEGV;
      logits[b * SEQ + s] = lg;
    }
  }
}

// q[b,h] = softmax(logits[b,:]) . ref[b,:,h]
__global__ __launch_bounds__(256) void kglimpse(
    const float* __restrict__ logits, const bf16_t* __restrict__ ref,
    float* __restrict__ q, bf16_t* __restrict__ qbf) {
  __shared__ float sp[SEQ];
  __shared__ float red[256];
  int b = blockIdx.x, tid = threadIdx.x;
  float lv = (tid < SEQ) ? logits[b * SEQ + tid] : -3.0e38f;
  red[tid] = lv; __syncthreads();
  for (int off = 128; off; off >>= 1) {
    if (tid < off) red[tid] = fmaxf(red[tid], red[tid + off]);
    __syncthreads();
  }
  float mx = red[0]; __syncthreads();
  float ev = (tid < SEQ) ? __expf(lv - mx) : 0.0f;
  red[tid] = ev; __syncthreads();
  for (int off = 128; off; off >>= 1) {
    if (tid < off) red[tid] += red[tid + off];
    __syncthreads();
  }
  float inv = 1.0f / red[0];
  if (tid < SEQ) sp[tid] = ev * inv;
  __syncthreads();
  float acc = 0.0f;                       // tid == output h index (H==256)
  const bf16_t* rp = ref + (size_t)b * SEQ * HID + tid;
  for (int s = 0; s < SEQ; ++s) acc += sp[s] * (float)rp[(size_t)s * HID];
  q[b * HID + tid] = acc;
  qbf[b * HID + tid] = (bf16_t)acc;
}

__device__ __forceinline__ unsigned hashu(unsigned x) {
  x ^= x >> 16; x *= 0x7feb352dU; x ^= x >> 15; x *= 0x846ca68bU; x ^= x >> 16;
  return x;
}

// Gumbel-max sample + log_softmax at chosen + mask update + din gather
__global__ __launch_bounds__(128) void ksample(
    const float* __restrict__ logits, int t, float* __restrict__ out,
    int* __restrict__ mask, const bf16_t* __restrict__ emb,
    bf16_t* __restrict__ din) {
  __shared__ float slog[SEQ];
  __shared__ float red[SEQ];
  __shared__ int   ridx[SEQ];
  __shared__ int   sch;
  int b = blockIdx.x, s = threadIdx.x;
  float lg = logits[b * SEQ + s];
  slog[s] = lg;
  red[s] = lg; __syncthreads();
  for (int off = 64; off; off >>= 1) {
    if (s < off) red[s] = fmaxf(red[s], red[s + off]);
    __syncthreads();
  }
  float mx = red[0]; __syncthreads();
  red[s] = __expf(lg - mx); __syncthreads();
  for (int off = 64; off; off >>= 1) {
    if (s < off) red[s] += red[s + off];
    __syncthreads();
  }
  float lse = mx + __logf(red[0]); __syncthreads();
  // deterministic per-(t,b,s) Gumbel noise
  unsigned u = hashu(0x9E3779B9u * (unsigned)t + 0x85EBCA6Bu * (unsigned)b + (unsigned)s + 1u);
  float uf = ((float)(u >> 8) + 0.5f) * (1.0f / 16777216.0f);
  float key = lg - __logf(-__logf(uf));
  red[s] = key; ridx[s] = s; __syncthreads();
  for (int off = 64; off; off >>= 1) {
    if (s < off && red[s + off] > red[s]) { red[s] = red[s + off]; ridx[s] = ridx[s + off]; }
    __syncthreads();
  }
  if (s == 0) {
    int ch = ridx[0]; sch = ch;
    out[(size_t)b * SEQ + t] = slog[ch] - lse;                        // logps [B,S]
    out[(size_t)BATCH * SEQ + (size_t)b * SEQ + t] = (float)ch;       // idxs  [B,S]
    mask[b * SEQ + ch] = 1;
  }
  __syncthreads();
  int ch = sch;                            // s == EMB index (EMB==128)
  din[b * EMB + s] = emb[((size_t)b * SEQ + ch) * EMB + s];
}

// ---------------------------------------------------------------------------
extern "C" void kernel_launch(void* const* d_in, const int* in_sizes, int n_in,
                              void* d_out, int out_size, void* d_ws, size_t ws_size,
                              hipStream_t stream) {
  (void)in_sizes; (void)n_in; (void)out_size; (void)ws_size;
  const float* x        = (const float*)d_in[0];
  const float* embed_w  = (const float*)d_in[1];
  const float* enc_Wih  = (const float*)d_in[2];
  const float* enc_Whh  = (const float*)d_in[3];
  const float* enc_bih  = (const float*)d_in[4];
  const float* enc_bhh  = (const float*)d_in[5];
  const float* dec_Wih  = (const float*)d_in[6];
  const float* dec_Whh  = (const float*)d_in[7];
  const float* dec_bih  = (const float*)d_in[8];
  const float* dec_bhh  = (const float*)d_in[9];
  const float* g_Wq     = (const float*)d_in[10];
  const float* g_bq     = (const float*)d_in[11];
  const float* g_Wr     = (const float*)d_in[12];
  const float* g_br     = (const float*)d_in[13];
  const float* g_V      = (const float*)d_in[14];
  const float* p_Wq     = (const float*)d_in[15];
  const float* p_bq     = (const float*)d_in[16];
  const float* p_Wr     = (const float*)d_in[17];
  const float* p_br     = (const float*)d_in[18];
  const float* p_V      = (const float*)d_in[19];
  const float* decstart = (const float*)d_in[20];
  float* out = (float*)d_out;

  char* ws = (char*)d_ws;
  size_t o = 0;
  auto take = [&](size_t bytes) -> void* {
    void* p = ws + o;
    o += (bytes + 255) & ~(size_t)255;
    return p;
  };
  bf16_t* Wenc   = (bf16_t*)take((size_t)G4 * KLSTM * 2);
  bf16_t* Wdec   = (bf16_t*)take((size_t)G4 * KLSTM * 2);
  float*  benc   = (float*) take((size_t)G4 * 4);
  float*  bdec   = (float*) take((size_t)G4 * 4);
  bf16_t* gWq    = (bf16_t*)take((size_t)HID * HID * 2);
  bf16_t* pWq    = (bf16_t*)take((size_t)HID * HID * 2);
  bf16_t* gWr    = (bf16_t*)take((size_t)HID * HID * 2);
  bf16_t* pWr    = (bf16_t*)take((size_t)HID * HID * 2);
  bf16_t* emb    = (bf16_t*)take((size_t)BATCH * SEQ * EMB * 2);
  bf16_t* encout = (bf16_t*)take((size_t)BATCH * SEQ * HID * 2);
  bf16_t* refg   = (bf16_t*)take((size_t)BATCH * SEQ * HID * 2);
  bf16_t* refp   = (bf16_t*)take((size_t)BATCH * SEQ * HID * 2);
  bf16_t* Abuf   = (bf16_t*)take((size_t)BATCH * KLSTM * 2);
  float*  gates  = (float*) take((size_t)BATCH * G4 * 4);
  float*  h      = (float*) take((size_t)BATCH * HID * 4);
  float*  c      = (float*) take((size_t)BATCH * HID * 4);
  bf16_t* hbf    = (bf16_t*)take((size_t)BATCH * HID * 2);
  bf16_t* din    = (bf16_t*)take((size_t)BATCH * EMB * 2);
  float*  ql     = (float*) take((size_t)BATCH * HID * 4);
  float*  q      = (float*) take((size_t)BATCH * HID * 4);
  bf16_t* qbf    = (bf16_t*)take((size_t)BATCH * HID * 2);
  float*  ql2    = (float*) take((size_t)BATCH * HID * 4);
  float*  logits = (float*) take((size_t)BATCH * SEQ * 4);
  int*    mask   = (int*)   take((size_t)BATCH * SEQ * 4);

  // ---- setup ----
  kcombine<<<(G4 * KLSTM + 255) / 256, 256, 0, stream>>>(enc_Wih, enc_Whh, Wenc);
  kcombine<<<(G4 * KLSTM + 255) / 256, 256, 0, stream>>>(dec_Wih, dec_Whh, Wdec);
  kbias2<<<(G4 + 255) / 256, 256, 0, stream>>>(enc_bih, enc_bhh, benc, G4);
  kbias2<<<(G4 + 255) / 256, 256, 0, stream>>>(dec_bih, dec_bhh, bdec, G4);
  kcvt<<<(HID * HID + 255) / 256, 256, 0, stream>>>(g_Wq, gWq, HID * HID);
  kcvt<<<(HID * HID + 255) / 256, 256, 0, stream>>>(p_Wq, pWq, HID * HID);
  kcvt<<<(HID * HID + 255) / 256, 256, 0, stream>>>(g_Wr, gWr, HID * HID);
  kcvt<<<(HID * HID + 255) / 256, 256, 0, stream>>>(p_Wr, pWr, HID * HID);
  kembed<<<(BATCH * SEQ * EMB + 255) / 256, 256, 0, stream>>>(x, embed_w, emb);
  kinit<<<(BATCH * HID + 255) / 256, 256, 0, stream>>>(h, c, din, mask, decstart);

  const dim3 blk(256);
  const dim3 gGate(G4 / 128, BATCH / 64);         // (8, 8)
  const dim3 gRef(HID / 128, (BATCH * SEQ) / 64); // (2, 1024)
  const dim3 gQ(HID / 128, BATCH / 64);           // (2, 8)

  // ---- encoder ----
  for (int t = 0; t < SEQ; ++t) {
    kpack<<<(BATCH * KLSTM + 255) / 256, 256, 0, stream>>>(emb, t, 1, h, Abuf);
    wmma_gemm_bf16<<<gGate, blk, 0, stream>>>(Abuf, Wenc, benc, gates, nullptr,
                                              BATCH, G4, KLSTM);
    klstm<<<(BATCH * HID + 255) / 256, 256, 0, stream>>>(gates, h, c, hbf, encout, t);
  }

  // ---- attention reference projections (once) ----
  wmma_gemm_bf16<<<gRef, blk, 0, stream>>>(encout, gWr, g_br, nullptr, refg,
                                           BATCH * SEQ, HID, HID);
  wmma_gemm_bf16<<<gRef, blk, 0, stream>>>(encout, pWr, p_br, nullptr, refp,
                                           BATCH * SEQ, HID, HID);

  // ---- decoder ----
  for (int t = 0; t < SEQ; ++t) {
    kpack<<<(BATCH * KLSTM + 255) / 256, 256, 0, stream>>>(din, t, 0, h, Abuf);
    wmma_gemm_bf16<<<gGate, blk, 0, stream>>>(Abuf, Wdec, bdec, gates, nullptr,
                                              BATCH, G4, KLSTM);
    klstm<<<(BATCH * HID + 255) / 256, 256, 0, stream>>>(gates, h, c, hbf, nullptr, t);
    // glimpse: ql = h @ g_Wq^T + g_bq
    wmma_gemm_bf16<<<gQ, blk, 0, stream>>>(hbf, gWq, g_bq, ql, nullptr,
                                           BATCH, HID, HID);
    kattlog<<<BATCH, 256, 0, stream>>>(ql, refg, g_V, mask, logits, 0);
    kglimpse<<<BATCH, 256, 0, stream>>>(logits, refg, q, qbf);
    // pointer: ql2 = q @ p_Wq^T + p_bq
    wmma_gemm_bf16<<<gQ, blk, 0, stream>>>(qbf, pWq, p_bq, ql2, nullptr,
                                           BATCH, HID, HID);
    kattlog<<<BATCH, 256, 0, stream>>>(ql2, refp, p_V, mask, logits, 1);
    ksample<<<BATCH, 128, 0, stream>>>(logits, t, out, mask, emb, din);
  }
}
